// Attention_12919261626862
// MI455X (gfx1250) — matrix operand order repaired and checked by reading, not compiled, not run on hardware
//
#include <hip/hip_runtime.h>

// ---------------------------------------------------------------------------
// CDNA5 (gfx1250) WMMA: wave32, v_wmma_f32_16x16x32_bf16
// ---------------------------------------------------------------------------
typedef __attribute__((ext_vector_type(16))) __bf16 v16bf;
typedef __attribute__((ext_vector_type(8)))  __bf16 v8bf;
typedef __attribute__((ext_vector_type(8)))  float  v8f;

#define B_  4
#define T_  2048
#define C_  1024
#define H_  16
#define D_  64

__device__ __forceinline__ v8f wmma_bf16(v16bf a, v16bf b, v8f c) {
    return __builtin_amdgcn_wmma_f32_16x16x32_bf16(
        false, a, false, b, (short)0, c, false, false);
}

// A-fragment (16x32 bf16): lane L -> row M = L%16, g = L/16,
// VGPR0..3 hold K = g*8+0..7, VGPR4..7 hold K = 16+g*8+0..7.
__device__ __forceinline__ v16bf load_a_frag(const __bf16* p, int ld) {
    const int lane = threadIdx.x & 31;
    const int r = lane & 15, g = lane >> 4;
    const __bf16* q = p + r * ld + g * 8;
    union { v16bf v; v8bf h[2]; } u;
    u.h[0] = *(const v8bf*)(q);
    u.h[1] = *(const v8bf*)(q + 16);
    return u.v;
}

// B-fragment (32x16 bf16): lane L -> col N = L%16, g = L/16,
// elements 0..7 hold K = g*8+0..7, elements 8..15 hold K = 16+g*8+0..7.
// `p` points at (K=0, N=0) of a K-major tile with row length `ld`.
__device__ __forceinline__ v16bf load_b_frag(const __bf16* p, int ld) {
    const int lane = threadIdx.x & 31;
    const int n = lane & 15, g = lane >> 4;
    v16bf b;
    #pragma unroll
    for (int s = 0; s < 8; ++s) {
        b[s]     = p[(g * 8 + s) * ld + n];
        b[8 + s] = p[(16 + g * 8 + s) * ld + n];
    }
    return b;
}

// ---------------------------------------------------------------------------
// 1) float -> bf16 convert (x)
// ---------------------------------------------------------------------------
__global__ __launch_bounds__(256) void k_f32_to_bf16(const float* __restrict__ in,
                                                     __bf16* __restrict__ out, int n) {
    int i = (blockIdx.x * 256 + threadIdx.x) * 4;
    if (i + 3 < n) {
        float4 f = *(const float4*)(in + i);
        out[i + 0] = (__bf16)f.x;
        out[i + 1] = (__bf16)f.y;
        out[i + 2] = (__bf16)f.z;
        out[i + 3] = (__bf16)f.w;
    }
}

// ---------------------------------------------------------------------------
// 2) QKV projection GEMM: Y[m,o] = sum_c x[m,c]*W[o,c].  M=8192,N=1024,K=1024.
//    Block 256 thr = 8 waves (4M x 2N); wave tile 32x64 (2x4 WMMA).
//    Block tile 128x128.  grid = (8, 64, 3), z selects Q/K/V.
//    Q,V stored (B,H,T,D) bf16;  K stored TRANSPOSED (B,H,D,T) bf16 so the
//    attention kernel's Q.K^T B-fragments are direct contiguous global loads.
// ---------------------------------------------------------------------------
__global__ __launch_bounds__(256) void k_gemm_qkv(
    const __bf16* __restrict__ xb,
    const float* __restrict__ Wq, const float* __restrict__ Wk, const float* __restrict__ Wv,
    __bf16* __restrict__ Qb, __bf16* __restrict__ Ktb, __bf16* __restrict__ Vb) {

    __shared__ __attribute__((aligned(32))) __bf16 ldsWT[32 * 128];  // [c][o] 8KB

    const int z = blockIdx.z;
    const float* W  = (z == 0) ? Wq : (z == 1) ? Wk : Wv;
    __bf16*     dst = (z == 0) ? Qb : (z == 1) ? Ktb : Vb;
    const bool  kt  = (z == 1);

    const int tid = threadIdx.x, w = tid >> 5, lane = tid & 31;
    const int r = lane & 15, g = lane >> 4;
    const int wm = w & 3, wn = w >> 2;
    const int mblk = blockIdx.y * 128, nblk = blockIdx.x * 128;
    const int mwave = mblk + wm * 32;
    const int nwave = wn * 64;

    v8f acc[2][4] = {};

    for (int kk = 0; kk < C_; kk += 32) {
        // stage W^T tile: ldsWT[c][o] = W[nblk+o][kk+c], f32->bf16
        {
            const int o_off = tid & 127;
            const int c16 = (tid >> 7) << 4;          // 0 or 16
            const float* ws = W + (nblk + o_off) * C_ + kk + c16;
            float4 f0 = *(const float4*)(ws);
            float4 f1 = *(const float4*)(ws + 4);
            float4 f2 = *(const float4*)(ws + 8);
            float4 f3 = *(const float4*)(ws + 12);
            float fv[16] = { f0.x, f0.y, f0.z, f0.w,  f1.x, f1.y, f1.z, f1.w,
                             f2.x, f2.y, f2.z, f2.w,  f3.x, f3.y, f3.z, f3.w };
            __bf16* dl = ldsWT + o_off;
            #pragma unroll
            for (int i = 0; i < 16; ++i) dl[(c16 + i) * 128] = (__bf16)fv[i];
        }
        __syncthreads();

        v16bf a0 = load_a_frag(xb + (mwave +  0) * C_ + kk, C_);
        v16bf a1 = load_a_frag(xb + (mwave + 16) * C_ + kk, C_);
        v16bf b[4];
        #pragma unroll
        for (int j = 0; j < 4; ++j)
            b[j] = load_b_frag(ldsWT + nwave + j * 16, 128);

        #pragma unroll
        for (int j = 0; j < 4; ++j) {
            acc[0][j] = wmma_bf16(a0, b[j], acc[0][j]);
            acc[1][j] = wmma_bf16(a1, b[j], acc[1][j]);
        }
        __syncthreads();
    }

    #pragma unroll
    for (int tm = 0; tm < 2; ++tm)
        #pragma unroll
        for (int j = 0; j < 4; ++j) {
            const int nt = nblk + nwave + j * 16;
            #pragma unroll
            for (int v = 0; v < 8; ++v) {
                const int m = mwave + tm * 16 + v + 8 * g;
                const int o = nt + r;
                const int bi = m >> 11, ti = m & 2047, hi = o >> 6, di = o & 63;
                const int bh = (bi << 4) | hi;
                const int idx = kt ? (bh * D_ + di) * T_ + ti      // (B,H,D,T)
                                   : (bh * T_ + ti) * D_ + di;     // (B,H,T,D)
                dst[idx] = (__bf16)acc[tm][j][v];
            }
        }
}

// ---------------------------------------------------------------------------
// 3) Flash attention (causal, online softmax).  grid = (T/256, B*H).
//    8 independent waves per block; each wave owns 32 query rows (2 q-tiles).
//    NO block barriers: K is consumed from the transposed (B,H,D,T) layout so
//    Q.K^T B-fragments are direct 32B global loads; V rows are contiguous.
//    Per 32-key block per wave: 4 K-frag + 4 V-frag loads, 16 WMMAs.
//    Each wave loops keys only to its own causal bound (exact trimming).
// ---------------------------------------------------------------------------
__global__ __launch_bounds__(256) void k_attn(
    const __bf16* __restrict__ Qb, const __bf16* __restrict__ Ktb,
    const __bf16* __restrict__ Vb, __bf16* __restrict__ Ob) {

    __shared__ __attribute__((aligned(32))) __bf16 ldsP[8 * 2 * 16 * 32]; // 16KB

    const int tid = threadIdx.x, w = tid >> 5, lane = tid & 31;
    const int r = lane & 15, g = lane >> 4;
    const int bh = blockIdx.y;
    const int q0 = blockIdx.x * 256 + w * 32;

    const __bf16* Qrow = Qb  + (bh * T_ + q0) * D_;
    const __bf16* Kt   = Ktb + bh * D_ * T_;          // [d][t]
    const __bf16* Vh   = Vb  + bh * T_ * D_;          // [t][d]
    __bf16* ldsPw = ldsP + w * (2 * 16 * 32);

    v16bf aq[2][2];
    #pragma unroll
    for (int t = 0; t < 2; ++t) {
        aq[t][0] = load_a_frag(Qrow + (t * 16) * D_ +  0, D_);
        aq[t][1] = load_a_frag(Qrow + (t * 16) * D_ + 32, D_);
    }

    v8f oacc[2][4] = {};
    float mrow[2][8], lrow[2][8];
    #pragma unroll
    for (int t = 0; t < 2; ++t)
        #pragma unroll
        for (int v = 0; v < 8; ++v) { mrow[t][v] = -__builtin_inff(); lrow[t][v] = 0.f; }

    const float scale = 0.125f;                       // 1/sqrt(64)
    const int kend = q0 + 32;                         // this wave's causal bound

    for (int kb = 0; kb < kend; kb += 32) {
        // K B-fragments: k-step s covers dims s*32..s*32+31; lane L holds key kb+n*16+L%16
        v16bf bk[2][2];
        #pragma unroll
        for (int s = 0; s < 2; ++s)
            #pragma unroll
            for (int n = 0; n < 2; ++n)
                bk[s][n] = load_b_frag(Kt + (s * 32) * T_ + kb + n * 16, T_);
        // V B-fragments: lane L = dim column j*16+L%16, keys kb..kb+31 along k
        v16bf bv[4];
        #pragma unroll
        for (int j = 0; j < 4; ++j)
            bv[j] = load_b_frag(Vh + kb * D_ + j * 16, D_);

        if (kb + 32 < kend) {                         // prefetch next key block
            __builtin_prefetch(Kt + lane * T_ + kb + 32, 0, 1);
            __builtin_prefetch(Vh + (kb + 32 + lane) * D_, 0, 1);
        }

        #pragma unroll
        for (int t = 0; t < 2; ++t) {
            v8f s0 = {}, s1 = {};
            s0 = wmma_bf16(aq[t][0], bk[0][0], s0);
            s0 = wmma_bf16(aq[t][1], bk[1][0], s0);
            s1 = wmma_bf16(aq[t][0], bk[0][1], s1);
            s1 = wmma_bf16(aq[t][1], bk[1][1], s1);

            #pragma unroll
            for (int v = 0; v < 8; ++v) {
                const int q = q0 + t * 16 + v + 8 * g;
                float e0 = s0[v] * scale; if (kb + r      > q) e0 = -__builtin_inff();
                float e1 = s1[v] * scale; if (kb + 16 + r > q) e1 = -__builtin_inff();
                float mx = fmaxf(e0, e1);
                #pragma unroll
                for (int sh = 1; sh < 16; sh <<= 1) mx = fmaxf(mx, __shfl_xor(mx, sh, 32));
                const float mnew  = fmaxf(mrow[t][v], mx);
                const float alpha = __expf(mrow[t][v] - mnew);
                e0 = __expf(e0 - mnew);
                e1 = __expf(e1 - mnew);
                float rs = e0 + e1;
                #pragma unroll
                for (int sh = 1; sh < 16; sh <<= 1) rs += __shfl_xor(rs, sh, 32);
                lrow[t][v] = lrow[t][v] * alpha + rs;
                mrow[t][v] = mnew;
                oacc[t][0][v] *= alpha; oacc[t][1][v] *= alpha;
                oacc[t][2][v] *= alpha; oacc[t][3][v] *= alpha;
                __bf16* pw = ldsPw + t * 512 + (v + 8 * g) * 32;
                pw[r]      = (__bf16)e0;
                pw[16 + r] = (__bf16)e1;
            }
            // intra-wave LDS RAW before A-layout reload of P
            asm volatile("s_wait_dscnt 0" ::: "memory");
            const v16bf pa = load_a_frag(ldsPw + t * 512, 32);
            #pragma unroll
            for (int j = 0; j < 4; ++j)
                oacc[t][j] = wmma_bf16(pa, bv[j], oacc[t][j]);
        }
    }

    // epilogue: normalize and write O in (B,T,H*D) bf16
    const int b = bh >> 4, h = bh & 15;
    #pragma unroll
    for (int t = 0; t < 2; ++t)
        #pragma unroll
        for (int j = 0; j < 4; ++j)
            #pragma unroll
            for (int v = 0; v < 8; ++v) {
                const int q = q0 + t * 16 + v + 8 * g;
                const float val = oacc[t][j][v] / lrow[t][v];
                Ob[(b * T_ + q) * C_ + h * D_ + j * 16 + r] = (__bf16)val;
            }
}

// ---------------------------------------------------------------------------
// 4) Output projection: out[m,o] = sum_c O[m,c]*W_out[o,c], fp32 store.
//    Same 128x128 / 32x64-wave geometry as the QKV GEMM.
// ---------------------------------------------------------------------------
__global__ __launch_bounds__(256) void k_gemm_out(
    const __bf16* __restrict__ Ob, const float* __restrict__ Wo,
    float* __restrict__ out) {

    __shared__ __attribute__((aligned(32))) __bf16 ldsWT[32 * 128];

    const int tid = threadIdx.x, w = tid >> 5, lane = tid & 31;
    const int r = lane & 15, g = lane >> 4;
    const int wm = w & 3, wn = w >> 2;
    const int mblk = blockIdx.y * 128, nblk = blockIdx.x * 128;
    const int mwave = mblk + wm * 32;
    const int nwave = wn * 64;

    v8f acc[2][4] = {};

    for (int kk = 0; kk < C_; kk += 32) {
        {
            const int o_off = tid & 127;
            const int c16 = (tid >> 7) << 4;
            const float* ws = Wo + (nblk + o_off) * C_ + kk + c16;
            float4 f0 = *(const float4*)(ws);
            float4 f1 = *(const float4*)(ws + 4);
            float4 f2 = *(const float4*)(ws + 8);
            float4 f3 = *(const float4*)(ws + 12);
            float fv[16] = { f0.x, f0.y, f0.z, f0.w,  f1.x, f1.y, f1.z, f1.w,
                             f2.x, f2.y, f2.z, f2.w,  f3.x, f3.y, f3.z, f3.w };
            __bf16* dl = ldsWT + o_off;
            #pragma unroll
            for (int i = 0; i < 16; ++i) dl[(c16 + i) * 128] = (__bf16)fv[i];
        }
        __syncthreads();

        v16bf a0 = load_a_frag(Ob + (mwave +  0) * C_ + kk, C_);
        v16bf a1 = load_a_frag(Ob + (mwave + 16) * C_ + kk, C_);
        v16bf b[4];
        #pragma unroll
        for (int j = 0; j < 4; ++j)
            b[j] = load_b_frag(ldsWT + nwave + j * 16, 128);

        #pragma unroll
        for (int j = 0; j < 4; ++j) {
            acc[0][j] = wmma_bf16(a0, b[j], acc[0][j]);
            acc[1][j] = wmma_bf16(a1, b[j], acc[1][j]);
        }
        __syncthreads();
    }

    #pragma unroll
    for (int tm = 0; tm < 2; ++tm)
        #pragma unroll
        for (int j = 0; j < 4; ++j) {
            const int nt = nblk + nwave + j * 16;
            #pragma unroll
            for (int v = 0; v < 8; ++v) {
                const int m = mwave + tm * 16 + v + 8 * g;
                out[m * C_ + nt + r] = acc[tm][j][v];
            }
        }
}

// ---------------------------------------------------------------------------
// Host launcher.  Workspace (80 MB):
//   xb | Qb (B,H,T,D) | Ktb (B,H,D,T) | Vb (B,H,T,D) | Ob (B,T,C)   all bf16
// ---------------------------------------------------------------------------
extern "C" void kernel_launch(void* const* d_in, const int* in_sizes, int n_in,
                              void* d_out, int out_size, void* d_ws, size_t ws_size,
                              hipStream_t stream) {
    (void)in_sizes; (void)n_in; (void)out_size; (void)ws_size;
    const float* x  = (const float*)d_in[0];
    const float* Wq = (const float*)d_in[1];
    const float* Wk = (const float*)d_in[2];
    const float* Wv = (const float*)d_in[3];
    const float* Wo = (const float*)d_in[4];
    float* out = (float*)d_out;

    const size_t NE = (size_t)B_ * T_ * C_;        // 8,388,608
    __bf16* xb  = (__bf16*)d_ws;
    __bf16* Qb  = xb  + NE;
    __bf16* Ktb = Qb  + NE;
    __bf16* Vb  = Ktb + NE;
    __bf16* Ob  = Vb  + NE;

    k_f32_to_bf16<<<(int)(NE / (256 * 4)), 256, 0, stream>>>(x, xb, (int)NE);

    dim3 g_qkv(C_ / 128, (B_ * T_) / 128, 3);
    k_gemm_qkv<<<g_qkv, 256, 0, stream>>>(xb, Wq, Wk, Wv, Qb, Ktb, Vb);

    dim3 g_att(T_ / 256, B_ * H_);
    k_attn<<<g_att, 256, 0, stream>>>(Qb, Ktb, Vb, Ob);

    dim3 g_out(C_ / 128, (B_ * T_) / 128);
    k_gemm_out<<<g_out, 256, 0, stream>>>(Ob, Wo, out);
}
